// GCN_61143154426422
// MI455X (gfx1250) — matrix operand order repaired and checked by reading, not compiled, not run on hardware
//
#include <hip/hip_runtime.h>

typedef __attribute__((ext_vector_type(16))) _Float16 v16h;
typedef __attribute__((ext_vector_type(8)))  float    v8f;
typedef __attribute__((ext_vector_type(4)))  float    v4f;

#define N_NODES 16384
#define DNBR    16
#define NGRAPH  32

union F16Frag { v16h v; v4f q[2]; };

__device__ __forceinline__ float fsig(float x) {
  return 1.0f / (1.0f + __builtin_amdgcn_exp2f(-1.4426950408889634f * x));
}
__device__ __forceinline__ float ftanh(float x) {
  return 2.0f * fsig(2.0f * x) - 1.0f;
}

// CDNA5 async global->LDS copy (tracked by ASYNCcnt). lds_off is a byte offset
// from the workgroup's dynamic-LDS base; per-lane src/dst, EXEC all-ones here.
__device__ __forceinline__ void g2lds_async_b128(unsigned lds_off, const _Float16* g) {
  unsigned long long ga = (unsigned long long)(size_t)g;
  asm volatile("global_load_async_to_lds_b128 %0, %1, off"
               :: "v"(lds_off), "v"(ga) : "memory");
}
__device__ __forceinline__ void wait_async0() {
  asm volatile("s_wait_asynccnt 0x0" ::: "memory");
}

__global__ void k_cvt(const float* __restrict__ s, _Float16* __restrict__ d, int n) {
  int i = blockIdx.x * blockDim.x + threadIdx.x;
  if (i < n) d[i] = (_Float16)s[i];
}

// d[c*R + r] = s[r*C + c]  (transpose + convert, for W_self / W_neigh)
__global__ void k_cvtT(const float* __restrict__ s, _Float16* __restrict__ d, int R, int C) {
  int i = blockIdx.x * blockDim.x + threadIdx.x;
  if (i < R * C) { int r = i / C, c = i - r * C; d[c * R + r] = (_Float16)s[i]; }
}

__device__ __forceinline__ void lstm_update8(
    const v8f& gi, const v8f& gf, const v8f& gg, const v8f& go,
    float bi, float bf, float bg, float bo,
    _Float16* c8,        // 8 contiguous f16 cell states (LDS)
    _Float16* hcol,      // -> hw[m0*FIN + col]; stride FIN per element
    int strideh)
{
  #pragma unroll
  for (int v = 0; v < 8; ++v) {
    float iv = fsig(gi[v] + bi);
    float fv = fsig(gf[v] + bf);
    float gv = ftanh(gg[v] + bg);
    float ov = fsig(go[v] + bo);
    float cn = fv * (float)c8[v] + iv * gv;
    c8[v] = (_Float16)cn;
    hcol[v * strideh] = (_Float16)(ov * ftanh(cn));
  }
}

// ---------------------------------------------------------------------------
// LSTM neighbor aggregator. One wave (== one workgroup, 32 lanes) owns a
// 32-node tile (two 16-row WMMA tiles -> every B fragment feeds 2 WMMAs).
// Per timestep: async-DMA the 32 gathered rows into LDS (ASYNCcnt), then
// gates = x_t@W_ih^T + h@W_hh^T via v_wmma_f32_16x16x32_f16, 16 cols at a time.
// h double-buffered f16 in LDS; c f16 in LDS.  LDS = 256*FIN bytes.
// ---------------------------------------------------------------------------
template <int FIN>
__global__ __launch_bounds__(32) void k_sage_lstm(
    const _Float16* __restrict__ hin,   // [N, FIN] layer input (f16)
    const int*      __restrict__ nbrs,  // [N, 16]
    const _Float16* __restrict__ Wih,   // [4H, FIN] row-major (== B^T)
    const _Float16* __restrict__ Whh,   // [4H, H]   row-major (== B^T)
    const float*    __restrict__ bih,   // [4H]
    const float*    __restrict__ bhh,   // [4H]
    _Float16*       __restrict__ hng)   // [N, FIN] h_neigh out (f16)
{
  constexpr int H  = FIN;
  constexpr int KT = FIN / 32;
  constexpr int NJ = H / 16;
  extern __shared__ char smem[];
  const int lane = threadIdx.x & 31;
  const int hl   = lane >> 4;
  const int r    = lane & 15;
  // LDS byte layout: [xt 64*FIN][hA 64*FIN][hB 64*FIN][c 64*FIN]
  _Float16* xt = (_Float16*)smem;                      // [32][FIN]
  _Float16* hA = (_Float16*)(smem + 64 * FIN);         // [32][FIN]
  _Float16* hB = (_Float16*)(smem + 128 * FIN);        // [32][FIN]
  _Float16* cc = (_Float16*)(smem + 192 * FIN);        // [col][32 rows]
  const int nodeBase = blockIdx.x * 32;

  for (int i = lane; i < 32 * FIN; i += 32) hA[i] = (_Float16)0.0f;
  for (int i = lane; i < 32 * H;  i += 32) cc[i] = (_Float16)0.0f;

  #pragma unroll 1
  for (int t = 0; t < DNBR; ++t) {
    _Float16* hr = (t & 1) ? hB : hA;
    _Float16* hw = (t & 1) ? hA : hB;

    // async gather: lane copies its whole row (FIN halfs) into LDS
    {
      const int nid = nbrs[(nodeBase + lane) * DNBR + t];
      const _Float16* src = hin + (size_t)nid * FIN;
      const unsigned dst0 = (unsigned)(lane * FIN * 2);
      #pragma unroll
      for (int i = 0; i < FIN / 8; ++i)
        g2lds_async_b128(dst0 + i * 16, src + i * 8);
      if (t + 1 < DNBR) {  // warm L2 for next timestep's gather
        int nn = nbrs[(nodeBase + lane) * DNBR + t + 1];
        __builtin_prefetch(hin + (size_t)nn * FIN, 0, 1);
      }
    }
    wait_async0();

    #pragma unroll 1
    for (int j = 0; j < NJ; ++j) {
      v8f acc0[4] = {};   // row tile 0 (rows 0..15)
      v8f acc1[4] = {};   // row tile 1 (rows 16..31)
      // gates += x_t @ W_ih^T  (A from LDS xt)
      #pragma unroll
      for (int kt = 0; kt < KT; ++kt) {
        F16Frag a0, a1;
        a0.q[0] = *(const v4f*)(xt + r * FIN + kt * 32 + hl * 8);
        a0.q[1] = *(const v4f*)(xt + r * FIN + kt * 32 + 16 + hl * 8);
        a1.q[0] = *(const v4f*)(xt + (16 + r) * FIN + kt * 32 + hl * 8);
        a1.q[1] = *(const v4f*)(xt + (16 + r) * FIN + kt * 32 + 16 + hl * 8);
        #pragma unroll
        for (int g = 0; g < 4; ++g) {
          F16Frag b;
          const _Float16* wp = Wih + (size_t)(g * H + j * 16 + r) * FIN + kt * 32 + hl * 8;
          b.q[0] = ((const v4f*)wp)[0];
          b.q[1] = ((const v4f*)wp)[2];
          acc0[g] = __builtin_amdgcn_wmma_f32_16x16x32_f16(false, a0.v, false, b.v,
                                                           (short)0, acc0[g], false, false);
          acc1[g] = __builtin_amdgcn_wmma_f32_16x16x32_f16(false, a1.v, false, b.v,
                                                           (short)0, acc1[g], false, false);
        }
      }
      // gates += h @ W_hh^T  (A from LDS h double buffer)
      #pragma unroll
      for (int kt = 0; kt < H / 32; ++kt) {
        F16Frag a0, a1;
        a0.q[0] = *(const v4f*)(hr + r * H + kt * 32 + hl * 8);
        a0.q[1] = *(const v4f*)(hr + r * H + kt * 32 + 16 + hl * 8);
        a1.q[0] = *(const v4f*)(hr + (16 + r) * H + kt * 32 + hl * 8);
        a1.q[1] = *(const v4f*)(hr + (16 + r) * H + kt * 32 + 16 + hl * 8);
        #pragma unroll
        for (int g = 0; g < 4; ++g) {
          F16Frag b;
          const _Float16* wp = Whh + (size_t)(g * H + j * 16 + r) * H + kt * 32 + hl * 8;
          b.q[0] = ((const v4f*)wp)[0];
          b.q[1] = ((const v4f*)wp)[2];
          acc0[g] = __builtin_amdgcn_wmma_f32_16x16x32_f16(false, a0.v, false, b.v,
                                                           (short)0, acc0[g], false, false);
          acc1[g] = __builtin_amdgcn_wmma_f32_16x16x32_f16(false, a1.v, false, b.v,
                                                           (short)0, acc1[g], false, false);
        }
      }
      // pointwise LSTM update; element v of tile T is row m = T*16+hl*8+v, col
      const int col = j * 16 + r;
      const float bi = bih[0 * H + col] + bhh[0 * H + col];
      const float bf = bih[1 * H + col] + bhh[1 * H + col];
      const float bg = bih[2 * H + col] + bhh[2 * H + col];
      const float bo = bih[3 * H + col] + bhh[3 * H + col];
      lstm_update8(acc0[0], acc0[1], acc0[2], acc0[3], bi, bf, bg, bo,
                   cc + col * 32 + hl * 8,
                   hw + (size_t)(hl * 8) * FIN + col, FIN);
      lstm_update8(acc1[0], acc1[1], acc1[2], acc1[3], bi, bf, bg, bo,
                   cc + col * 32 + 16 + hl * 8,
                   hw + (size_t)(16 + hl * 8) * FIN + col, FIN);
    }
  }
  // DNBR even -> final hidden lives in hA; lane streams out its row
  {
    const v4f* src = (const v4f*)(hA + lane * FIN);
    v4f* dst = (v4f*)(hng + (size_t)(nodeBase + lane) * FIN);
    #pragma unroll
    for (int i = 0; i < FIN / 8; ++i) dst[i] = src[i];
  }
}

// ---------------------------------------------------------------------------
// h_out = relu(h_self @ W_self + h_neigh @ W_neigh + b).  32 rows per wave:
// each weight fragment feeds two WMMAs.
// ---------------------------------------------------------------------------
template <int FIN, int FOUT>
__global__ __launch_bounds__(128) void k_sage_proj(
    const _Float16* __restrict__ hself,  // [N, FIN]
    const _Float16* __restrict__ hng,    // [N, FIN]
    const _Float16* __restrict__ WsT,    // [FOUT, FIN] (pre-transposed)
    const _Float16* __restrict__ WnT,    // [FOUT, FIN]
    const float*    __restrict__ bias,   // [FOUT]
    _Float16*       __restrict__ hout)   // [N, FOUT]
{
  const int wave = threadIdx.x >> 5;
  const int lane = threadIdx.x & 31;
  const int hl = lane >> 4, r = lane & 15;
  const int rowBase = (blockIdx.x * 4 + wave) * 32;
  const _Float16* s0 = hself + (size_t)(rowBase + r) * FIN;
  const _Float16* s1 = hself + (size_t)(rowBase + 16 + r) * FIN;
  const _Float16* n0 = hng + (size_t)(rowBase + r) * FIN;
  const _Float16* n1 = hng + (size_t)(rowBase + 16 + r) * FIN;
  #pragma unroll 1
  for (int jn = 0; jn < FOUT / 16; ++jn) {
    v8f acc0 = {}, acc1 = {};
    #pragma unroll
    for (int kt = 0; kt < FIN / 32; ++kt) {
      F16Frag a0, a1, b;
      a0.q[0] = *(const v4f*)(s0 + kt * 32 + hl * 8);
      a0.q[1] = *(const v4f*)(s0 + kt * 32 + 16 + hl * 8);
      a1.q[0] = *(const v4f*)(s1 + kt * 32 + hl * 8);
      a1.q[1] = *(const v4f*)(s1 + kt * 32 + 16 + hl * 8);
      const _Float16* wp = WsT + (size_t)(jn * 16 + r) * FIN + kt * 32 + hl * 8;
      b.q[0] = ((const v4f*)wp)[0];
      b.q[1] = ((const v4f*)wp)[2];
      acc0 = __builtin_amdgcn_wmma_f32_16x16x32_f16(false, a0.v, false, b.v,
                                                    (short)0, acc0, false, false);
      acc1 = __builtin_amdgcn_wmma_f32_16x16x32_f16(false, a1.v, false, b.v,
                                                    (short)0, acc1, false, false);
      a0.q[0] = *(const v4f*)(n0 + kt * 32 + hl * 8);
      a0.q[1] = *(const v4f*)(n0 + kt * 32 + 16 + hl * 8);
      a1.q[0] = *(const v4f*)(n1 + kt * 32 + hl * 8);
      a1.q[1] = *(const v4f*)(n1 + kt * 32 + 16 + hl * 8);
      wp = WnT + (size_t)(jn * 16 + r) * FIN + kt * 32 + hl * 8;
      b.q[0] = ((const v4f*)wp)[0];
      b.q[1] = ((const v4f*)wp)[2];
      acc0 = __builtin_amdgcn_wmma_f32_16x16x32_f16(false, a0.v, false, b.v,
                                                    (short)0, acc0, false, false);
      acc1 = __builtin_amdgcn_wmma_f32_16x16x32_f16(false, a1.v, false, b.v,
                                                    (short)0, acc1, false, false);
    }
    const float bv = bias[jn * 16 + r];
    #pragma unroll
    for (int v = 0; v < 8; ++v) {
      float y0 = acc0[v] + bv;
      float y1 = acc1[v] + bv;
      y0 = y0 > 0.0f ? y0 : 0.0f;
      y1 = y1 > 0.0f ? y1 : 0.0f;
      hout[(size_t)(rowBase + hl * 8 + v) * FOUT + jn * 16 + r] = (_Float16)y0;
      hout[(size_t)(rowBase + 16 + hl * 8 + v) * FOUT + jn * 16 + r] = (_Float16)y1;
    }
  }
}

// ---------------------------------------------------------------------------
// Per-graph segment_max over contiguous 512-node segments + fc1(relu)+fc2(relu)
// ---------------------------------------------------------------------------
__global__ __launch_bounds__(256) void k_pool_fc(
    const _Float16* __restrict__ h,  // [N, 16]
    const float* __restrict__ fc1w, const float* __restrict__ fc1b,
    const float* __restrict__ fc2w, const float* __restrict__ fc2b,
    float* __restrict__ out)         // [B]
{
  __shared__ float red[256 * 16];
  const int g = blockIdx.x, tid = threadIdx.x;
  float m[16];
  #pragma unroll
  for (int k = 0; k < 16; ++k) m[k] = -3.0e38f;
  for (int n = tid; n < 512; n += 256) {
    const _Float16* row = h + (size_t)(g * 512 + n) * 16;
    #pragma unroll
    for (int k = 0; k < 16; ++k) {
      float v = (float)row[k];
      m[k] = v > m[k] ? v : m[k];
    }
  }
  #pragma unroll
  for (int k = 0; k < 16; ++k) red[tid * 16 + k] = m[k];
  __syncthreads();
  for (int s = 128; s > 0; s >>= 1) {
    if (tid < s) {
      #pragma unroll
      for (int k = 0; k < 16; ++k) {
        float a = red[tid * 16 + k], b = red[(tid + s) * 16 + k];
        red[tid * 16 + k] = a > b ? a : b;
      }
    }
    __syncthreads();
  }
  if (tid == 0) {
    float z[8];
    #pragma unroll
    for (int j = 0; j < 8; ++j) {
      float s = fc1b[j];
      #pragma unroll
      for (int k = 0; k < 16; ++k) s += red[k] * fc1w[k * 8 + j];
      z[j] = s > 0.0f ? s : 0.0f;
    }
    float o = fc2b[0];
    #pragma unroll
    for (int j = 0; j < 8; ++j) o += z[j] * fc2w[j];
    out[g] = o > 0.0f ? o : 0.0f;
  }
}

extern "C" void kernel_launch(void* const* d_in, const int* in_sizes, int n_in,
                              void* d_out, int out_size, void* d_ws, size_t ws_size,
                              hipStream_t stream) {
  (void)n_in; (void)out_size; (void)ws_size;
  const float* x = (const float*)d_in[0];
  const float *Wih[5], *Whh[5], *Wself[5], *Wneigh[5], *bL[5], *Bih[5], *Bhh[5];
  const float *fc1w, *fc1b, *fc2w, *fc2b;
  // Two plausible flattenings of the params pytree; detect by in_sizes[1]:
  // sorted-key (jax pytree) order puts fc1_b (8 elems) at index 1.
  if (in_sizes[1] == 8) {
    fc1b = (const float*)d_in[1]; fc1w = (const float*)d_in[2];
    fc2b = (const float*)d_in[3]; fc2w = (const float*)d_in[4];
    for (int i = 0; i < 5; ++i) {  // per-layer sorted: W_hh,W_ih,W_neigh,W_self,b,b_hh,b_ih
      int k = 5 + 7 * i;
      Whh[i]    = (const float*)d_in[k + 0];
      Wih[i]    = (const float*)d_in[k + 1];
      Wneigh[i] = (const float*)d_in[k + 2];
      Wself[i]  = (const float*)d_in[k + 3];
      bL[i]     = (const float*)d_in[k + 4];
      Bhh[i]    = (const float*)d_in[k + 5];
      Bih[i]    = (const float*)d_in[k + 6];
    }
  } else {  // insertion order: layers first, each W_ih,W_hh,b_ih,b_hh,W_self,W_neigh,b
    for (int i = 0; i < 5; ++i) {
      int k = 1 + 7 * i;
      Wih[i]    = (const float*)d_in[k + 0];
      Whh[i]    = (const float*)d_in[k + 1];
      Bih[i]    = (const float*)d_in[k + 2];
      Bhh[i]    = (const float*)d_in[k + 3];
      Wself[i]  = (const float*)d_in[k + 4];
      Wneigh[i] = (const float*)d_in[k + 5];
      bL[i]     = (const float*)d_in[k + 6];
    }
    fc1w = (const float*)d_in[36]; fc1b = (const float*)d_in[37];
    fc2w = (const float*)d_in[38]; fc2b = (const float*)d_in[39];
  }
  const int* nbrs = (const int*)d_in[40];  // graph_id (d_in[41]) unused: segments contiguous

  // workspace carve-out
  char* wsp = (char*)d_ws;
  size_t off = 0;
  auto carve = [&](size_t bytes) -> char* {
    char* p = wsp + off;
    off += (bytes + 255) & ~(size_t)255;
    return p;
  };
  _Float16* bufA = (_Float16*)carve((size_t)N_NODES * 256 * 2);
  _Float16* bufB = (_Float16*)carve((size_t)N_NODES * 256 * 2);
  _Float16* bufN = (_Float16*)carve((size_t)N_NODES * 256 * 2);
  static const int fin_[5]  = {128, 256, 128, 64, 32};
  static const int fout_[5] = {256, 128, 64, 32, 16};
  _Float16 *Wih16[5], *Whh16[5], *WsT16[5], *WnT16[5];
  for (int i = 0; i < 5; ++i) {
    int H = fin_[i], G = 4 * H;
    Wih16[i] = (_Float16*)carve((size_t)G * H * 2);
    Whh16[i] = (_Float16*)carve((size_t)G * H * 2);
    WsT16[i] = (_Float16*)carve((size_t)fin_[i] * fout_[i] * 2);
    WnT16[i] = (_Float16*)carve((size_t)fin_[i] * fout_[i] * 2);
  }

  auto cvt = [&](const float* s, _Float16* d, int n) {
    k_cvt<<<dim3((n + 255) / 256), dim3(256), 0, stream>>>(s, d, n);
  };
  cvt(x, bufA, N_NODES * 128);
  for (int i = 0; i < 5; ++i) {
    int H = fin_[i], G = 4 * H, nt = fin_[i] * fout_[i];
    cvt(Wih[i], Wih16[i], G * H);
    cvt(Whh[i], Whh16[i], G * H);
    k_cvtT<<<dim3((nt + 255) / 256), dim3(256), 0, stream>>>(Wself[i],  WsT16[i], fin_[i], fout_[i]);
    k_cvtT<<<dim3((nt + 255) / 256), dim3(256), 0, stream>>>(Wneigh[i], WnT16[i], fin_[i], fout_[i]);
  }

  _Float16* hin = bufA;
  _Float16* hout = bufB;
#define RUN_LAYER(I, FI, FO)                                                     \
  k_sage_lstm<FI><<<dim3(N_NODES / 32), dim3(32), 256 * FI, stream>>>(           \
      hin, nbrs, Wih16[I], Whh16[I], Bih[I], Bhh[I], bufN);                      \
  k_sage_proj<FI, FO><<<dim3(N_NODES / 128), dim3(128), 0, stream>>>(            \
      hin, bufN, WsT16[I], WnT16[I], bL[I], hout);                               \
  { _Float16* tswap = hin; hin = hout; hout = tswap; }

  RUN_LAYER(0, 128, 256)
  RUN_LAYER(1, 256, 128)
  RUN_LAYER(2, 128, 64)
  RUN_LAYER(3, 64, 32)
  RUN_LAYER(4, 32, 16)
#undef RUN_LAYER

  k_pool_fc<<<dim3(NGRAPH), dim3(256), 0, stream>>>(hin, fc1w, fc1b, fc2w, fc2b,
                                                    (float*)d_out);
}
